// DiffusionPolicy_78838419685784
// MI455X (gfx1250) — matrix operand order repaired and checked by reading, not compiled, not run on hardware
//
#include <hip/hip_runtime.h>
#include <hip/hip_bf16.h>

typedef __attribute__((ext_vector_type(16))) __bf16 v16bf;
typedef __attribute__((ext_vector_type(8)))  float  v8f;

#define TPB 128

// ---- shared memory layout -------------------------------------------------
#define SM_F32_HDR 164                 // 32 x + 32 eps + 5*20 schedule floats
#define OFF_ACT   0                    // bf16 offsets after the f32 header
#define OFF_COND  (OFF_ACT  + 16*64)
#define OFF_HA    (OFF_COND + 16*128)
#define OFF_HB    (OFF_HA   + 16*256)
#define OFF_SC    (OFF_HB   + 16*256)
#define OFF_SH    (OFF_SC   + 4*16*256)
#define BF_TOTAL  (OFF_SH   + 4*16*256)
#define SMEM_BYTES (SM_F32_HDR*4 + BF_TOTAL*2)   // 88720 bytes < 320KB/WGP

// ---- helpers --------------------------------------------------------------
union V16Q { uint4 q[2]; v16bf v; };

__device__ __forceinline__ float siluf(float x) {
  // x * sigmoid(x); denominator in [1,inf) -> hardware v_rcp_f32 is safe
  return x * __builtin_amdgcn_rcpf(1.0f + __expf(-x));
}

__device__ __forceinline__ unsigned hashu(unsigned x) {
  x ^= x >> 17; x *= 0xed5ad4bbu;
  x ^= x >> 11; x *= 0xac4c1b51u;
  x ^= x >> 15; x *= 0x31848babu;
  x ^= x >> 14; return x;
}

__device__ __forceinline__ float gaussf(unsigned seed) {
  unsigned h1 = hashu(seed * 0x9e3779b9u + 0x85ebca6bu);
  unsigned h2 = hashu(h1 + 0xc2b2ae35u);
  float u1 = (float)(h1 >> 8) * (1.0f / 16777216.0f) + 1e-7f;
  float u2 = (float)(h2 >> 8) * (1.0f / 16777216.0f);
  return sqrtf(-2.0f * __logf(u1)) * __cosf(6.28318530718f * u2);
}

// A fragment: 16x32 bf16 tile, row-major in LDS.
// lanes 0-15 -> K 0..7 / 16..23, lanes 16-31 -> K 8..15 / 24..31
// Each lane: two contiguous 16B runs -> 2x ds_load_b128.
__device__ __forceinline__ v16bf load_a_lds(const __bf16* base, int ldk, int lane) {
  const int m  = lane & 15;
  const int kh = (lane >> 4) * 8;
  const __bf16* row = base + m * ldk + kh;
  V16Q t;
  t.q[0] = *(const uint4*)(row);        // K kh..kh+7
  t.q[1] = *(const uint4*)(row + 16);   // K kh+16..kh+23
  return t.v;
}

// B fragment: 32x16 bf16 from transposed weights WT[n][k].
// lanes 0-15 -> K 0..7 / 16..23, lanes 16-31 -> K 8..15 / 24..31; lane%16 selects column.
// Each lane: two contiguous 16B runs -> 2x global_load_b128.
__device__ __forceinline__ v16bf load_b_g(const __bf16* wt, int ldk, int lane) {
  const int n  = lane & 15;
  const int kh = (lane >> 4) * 8;
  const __bf16* col = wt + n * ldk + kh;
  V16Q t;
  t.q[0] = *(const uint4*)(col);
  t.q[1] = *(const uint4*)(col + 16);
  return t.v;
}

__device__ __forceinline__ v8f wmma_bf16(v16bf a, v16bf b, v8f c) {
  return __builtin_amdgcn_wmma_f32_16x16x32_bf16(
      /*neg_a=*/false, a, /*neg_b=*/false, b,
      /*c_mod=*/(short)0, c, /*reuse_a=*/false, /*reuse_b=*/false);
}

// ---- weight pack kernel: f32 [b][K][N] -> bf16 transposed padded [b][Np][Kp]
__global__ void pack_w_kernel(const float* __restrict__ src, __bf16* __restrict__ dst,
                              int K, int Nc, int Kp, int Np, int batch) {
  const long per = (long)Np * Kp;
  const long total = per * batch;
  for (long idx = blockIdx.x * (long)blockDim.x + threadIdx.x; idx < total;
       idx += (long)gridDim.x * blockDim.x) {
    int b   = (int)(idx / per);
    int rem = (int)(idx - (long)b * per);
    int n = rem / Kp;
    int k = rem - n * Kp;
    float v = (k < K && n < Nc) ? src[((long)b * K + k) * Nc + n] : 0.0f;
    dst[idx] = (__bf16)v;
  }
}

// ---- main persistent-per-tile diffusion kernel ----------------------------
__global__ __launch_bounds__(TPB) void diffusion_policy_kernel(
    const float* __restrict__ states,
    const float* __restrict__ W_se1, const float* __restrict__ b_se1,
    const float* __restrict__ b_se2, const float* __restrict__ b_in,
    const float* __restrict__ b_hid, const float* __restrict__ b_scale,
    const float* __restrict__ b_shift, const float* __restrict__ b_out,
    const __bf16* __restrict__ WinT,  const __bf16* __restrict__ WhidT,
    const __bf16* __restrict__ WscaleT, const __bf16* __restrict__ WshiftT,
    const __bf16* __restrict__ Wse2T, const __bf16* __restrict__ WoutT,
    const int* __restrict__ nsp, float* __restrict__ out, int N) {

  extern __shared__ char smem_raw[];
  float* fhdr   = (float*)smem_raw;
  float* xbuf   = fhdr;          // [16][2]
  float* epsb   = fhdr + 32;     // [16][2]
  float* s_src  = fhdr + 64;     // [20]
  float* s_srm1 = fhdr + 84;
  float* s_c1   = fhdr + 104;
  float* s_c2   = fhdr + 124;
  float* s_plv  = fhdr + 144;
  __bf16* bfb    = (__bf16*)(fhdr + SM_F32_HDR);
  __bf16* actIn  = bfb + OFF_ACT;   // [16][64]  (x | sin | cos | zero pad)
  __bf16* cond   = bfb + OFF_COND;  // [16][128]
  __bf16* hA     = bfb + OFF_HA;    // [16][256]
  __bf16* hB     = bfb + OFF_HB;    // [16][256]
  __bf16* scaleS = bfb + OFF_SC;    // [4][16][256]  step-invariant FiLM scale
  __bf16* shiftS = bfb + OFF_SH;    // [4][16][256]  step-invariant FiLM shift

  const int tid   = threadIdx.x;
  const int lane  = tid & 31;
  const int wv    = tid >> 5;           // wave id 0..3, each owns 4 col tiles
  const int nIdx  = lane & 15;
  const int mBase = (lane >> 4) * 8;    // C-tile row base per ISA layout
  const int gRow0 = blockIdx.x * 16;
  const v8f z = {};

  const int ns = *nsp;

  // ---- schedule (thread 0, sequential cumprod; trivial) ----
  if (tid == 0) {
    float run = 1.0f;
    for (int t2 = 0; t2 < 20; ++t2) {
      float beta  = 1e-4f + (0.02f - 1e-4f) * (float)t2 / 19.0f;
      float alpha = 1.0f - beta;
      float acp   = run;      // alphas_cumprod_prev
      run *= alpha;           // alphas_cumprod
      float one_m = fmaxf(1.0f - run, 1e-8f);
      s_src[t2]  = sqrtf(1.0f / run);
      s_srm1[t2] = sqrtf(1.0f / run - 1.0f);
      s_c1[t2]   = beta * sqrtf(acp) / one_m;
      s_c2[t2]   = (1.0f - acp) * sqrtf(alpha) / one_m;
      s_plv[t2]  = __logf(fmaxf(beta * (1.0f - acp) / one_m, 1e-20f));
    }
  }

  // ---- init x0 ~ N(0,1) ----
  if (tid < 32) {
    int r = tid >> 1, cc = tid & 1;
    unsigned gr = (unsigned)(gRow0 + r);
    xbuf[tid] = gaussf((gr * 2u + (unsigned)cc) ^ 0xA511E9B3u);
  }

  // ---- cond layer 1 (K=2, VALU) into hA scratch [16][128] ----
  for (int idx = tid; idx < 16 * 128; idx += TPB) {
    int r = idx >> 7, j = idx & 127;
    int gr = gRow0 + r; if (gr > N - 1) gr = N - 1;
    int si = gr / ns;
    float x0 = states[si * 2 + 0], x1 = states[si * 2 + 1];
    float v = x0 * W_se1[j] + x1 * W_se1[128 + j] + b_se1[j];
    hA[idx] = (__bf16)siluf(v);
  }
  __syncthreads();

  // ---- cond = silu(hA @ W_se2 + b_se2)  [16,128]x[128,128], WMMA ----
  {
    v8f acc[2] = {z, z};
#pragma unroll
    for (int ks = 0; ks < 4; ++ks) {
      v16bf a = load_a_lds(hA + ks * 32, 128, lane);
#pragma unroll
      for (int tt = 0; tt < 2; ++tt) {
        int cb = (wv + 4 * tt) * 16;
        v16bf b = load_b_g(Wse2T + cb * 128 + ks * 32, 128, lane);
        acc[tt] = wmma_bf16(a, b, acc[tt]);
      }
    }
    __syncthreads();   // all hA reads complete before reuse ordering below
#pragma unroll
    for (int tt = 0; tt < 2; ++tt) {
      int n = (wv + 4 * tt) * 16 + nIdx;
      float bias = b_se2[n];
#pragma unroll
      for (int j = 0; j < 8; ++j)
        cond[(mBase + j) * 128 + n] = (__bf16)siluf(acc[tt][j] + bias);
    }
  }
  __syncthreads();

  // ---- step-invariant FiLM: scale/shift per layer, LDS-resident (WMMA) ----
  for (int L = 0; L < 4; ++L) {
    v8f accS[4] = {z, z, z, z};
    v8f accH[4] = {z, z, z, z};
#pragma unroll
    for (int ks = 0; ks < 4; ++ks) {
      v16bf a = load_a_lds(cond + ks * 32, 128, lane);
#pragma unroll
      for (int tt = 0; tt < 4; ++tt) {
        int cb = (wv + 4 * tt) * 16;
        v16bf bs = load_b_g(WscaleT + (L * 256 + cb) * 128 + ks * 32, 128, lane);
        accS[tt] = wmma_bf16(a, bs, accS[tt]);
        v16bf bh = load_b_g(WshiftT + (L * 256 + cb) * 128 + ks * 32, 128, lane);
        accH[tt] = wmma_bf16(a, bh, accH[tt]);
      }
    }
#pragma unroll
    for (int tt = 0; tt < 4; ++tt) {
      int n = (wv + 4 * tt) * 16 + nIdx;
      float bs = b_scale[L * 256 + n];
      float bh = b_shift[L * 256 + n];
#pragma unroll
      for (int j = 0; j < 8; ++j) {
        int m = mBase + j;
        scaleS[(L * 16 + m) * 256 + n] = (__bf16)(1.0f + accS[tt][j] + bs);
        shiftS[(L * 16 + m) * 256 + n] = (__bf16)(accH[tt][j] + bh);
      }
    }
  }
  __syncthreads();

  // ---- 20 sequential denoise steps, all local to this workgroup ----
  for (int s = 0; s < 20; ++s) {
    const int t = 19 - s;
    const float tf = (float)t;

    // build actIn = [x0,x1, sin(t*f), cos(t*f), 0-pad] as bf16 [16][64]
    for (int idx = tid; idx < 16 * 64; idx += TPB) {
      int r = idx >> 6, c = idx & 63;
      float v;
      if (c < 2)       v = xbuf[r * 2 + c];
      else if (c < 18) v = __sinf(tf * __expf((float)(c - 2)  * 0.46051701860f));
      else if (c < 34) v = __cosf(tf * __expf((float)(c - 18) * 0.46051701860f));
      else             v = 0.0f;
      actIn[idx] = (__bf16)v;
    }
    __syncthreads();

    // input layer: hA = actIn @ W_in + b_in   (K=64 padded)
    {
      v8f acc[4] = {z, z, z, z};
#pragma unroll
      for (int ks = 0; ks < 2; ++ks) {
        v16bf a = load_a_lds(actIn + ks * 32, 64, lane);
#pragma unroll
        for (int tt = 0; tt < 4; ++tt) {
          int cb = (wv + 4 * tt) * 16;
          v16bf b = load_b_g(WinT + cb * 64 + ks * 32, 64, lane);
          acc[tt] = wmma_bf16(a, b, acc[tt]);
        }
      }
#pragma unroll
      for (int tt = 0; tt < 4; ++tt) {
        int n = (wv + 4 * tt) * 16 + nIdx;
        float bias = b_in[n];
#pragma unroll
        for (int j = 0; j < 8; ++j)
          hA[(mBase + j) * 256 + n] = (__bf16)(acc[tt][j] + bias);
      }
    }
    __syncthreads();

    // 4 hidden layers: h = silu((h@W + b) * scale + shift), ping-pong hA/hB
    for (int L = 0; L < 4; ++L) {
      __bf16* src = (L & 1) ? hB : hA;
      __bf16* dst = (L & 1) ? hA : hB;
      v8f acc[4] = {z, z, z, z};
#pragma unroll
      for (int ks = 0; ks < 8; ++ks) {
        v16bf a = load_a_lds(src + ks * 32, 256, lane);
#pragma unroll
        for (int tt = 0; tt < 4; ++tt) {
          int cb = (wv + 4 * tt) * 16;
          v16bf b = load_b_g(WhidT + (L * 256 + cb) * 256 + ks * 32, 256, lane);
          acc[tt] = wmma_bf16(a, b, acc[tt]);
        }
      }
#pragma unroll
      for (int tt = 0; tt < 4; ++tt) {
        int n = (wv + 4 * tt) * 16 + nIdx;
        float bias = b_hid[L * 256 + n];
#pragma unroll
        for (int j = 0; j < 8; ++j) {
          int m = mBase + j;
          float h = acc[tt][j] + bias;
          float sc = (float)scaleS[(L * 16 + m) * 256 + n];
          float sh = (float)shiftS[(L * 16 + m) * 256 + n];
          dst[m * 256 + n] = (__bf16)siluf(h * sc + sh);
        }
      }
      __syncthreads();
    }

    // output: eps = hA @ W_out + b_out  (N padded 2->16, wave 0 only)
    if (wv == 0) {
      v8f acc = z;
#pragma unroll
      for (int ks = 0; ks < 8; ++ks) {
        v16bf a = load_a_lds(hA + ks * 32, 256, lane);
        v16bf b = load_b_g(WoutT + ks * 32, 256, lane);
        acc = wmma_bf16(a, b, acc);
      }
      if (nIdx < 2) {
#pragma unroll
        for (int j = 0; j < 8; ++j)
          epsb[(mBase + j) * 2 + nIdx] = acc[j] + b_out[nIdx];
      }
    }
    __syncthreads();

    // DDPM posterior update per row element
    if (tid < 32) {
      int r = tid >> 1, cc = tid & 1;
      float x = xbuf[tid];
      float e = epsb[tid];
      float x0p  = fminf(fmaxf(s_src[t] * x - s_srm1[t] * e, -1.0f), 1.0f);
      float mean = s_c1[t] * x0p + s_c2[t] * x;
      float sigma = (t > 0) ? __expf(0.5f * s_plv[t]) : 0.0f;
      unsigned gr = (unsigned)(gRow0 + r);
      float nz = gaussf(((unsigned)t * 1315423911u) ^ (gr * 2654435761u + (unsigned)cc));
      xbuf[tid] = fminf(fmaxf(mean + sigma * nz, -1.5f), 1.5f);
    }
    __syncthreads();
  }

  // ---- write result (DATA_SCALE == 1.0) ----
  if (tid < 32) {
    int r = tid >> 1;
    int gr = gRow0 + r;
    if (gr < N) out[gr * 2 + (tid & 1)] = xbuf[tid];
  }
}

// ---- launcher -------------------------------------------------------------
extern "C" void kernel_launch(void* const* d_in, const int* in_sizes, int n_in,
                              void* d_out, int out_size, void* d_ws, size_t ws_size,
                              hipStream_t stream) {
  const float* states  = (const float*)d_in[0];
  const float* W_se1   = (const float*)d_in[1];
  const float* b_se1   = (const float*)d_in[2];
  const float* W_se2   = (const float*)d_in[3];
  const float* b_se2   = (const float*)d_in[4];
  const float* W_in    = (const float*)d_in[5];
  const float* b_in    = (const float*)d_in[6];
  const float* W_hid   = (const float*)d_in[7];
  const float* b_hid   = (const float*)d_in[8];
  const float* W_scale = (const float*)d_in[9];
  const float* b_scale = (const float*)d_in[10];
  const float* W_shift = (const float*)d_in[11];
  const float* b_shift = (const float*)d_in[12];
  const float* W_out   = (const float*)d_in[13];
  const float* b_out   = (const float*)d_in[14];
  const int*   nsp     = (const int*)d_in[15];
  float* out = (float*)d_out;
  const int N = out_size / 2;

  // workspace: bf16 transposed+padded weights (all L2-resident, ~1.1 MB)
  __bf16* ws      = (__bf16*)d_ws;
  __bf16* WinT    = ws;                          // [256][64]
  __bf16* WhidT   = WinT    + 256 * 64;          // [4][256][256]
  __bf16* WscaleT = WhidT   + 4 * 256 * 256;     // [4][256][128]
  __bf16* WshiftT = WscaleT + 4 * 256 * 128;     // [4][256][128]
  __bf16* Wse2T   = WshiftT + 4 * 256 * 128;     // [128][128]
  __bf16* WoutT   = Wse2T   + 128 * 128;         // [16][256]

  pack_w_kernel<<<64,  256, 0, stream>>>(W_in,    WinT,    34, 256,  64, 256, 1);
  pack_w_kernel<<<256, 256, 0, stream>>>(W_hid,   WhidT,  256, 256, 256, 256, 4);
  pack_w_kernel<<<128, 256, 0, stream>>>(W_scale, WscaleT,128, 256, 128, 256, 4);
  pack_w_kernel<<<128, 256, 0, stream>>>(W_shift, WshiftT,128, 256, 128, 256, 4);
  pack_w_kernel<<<16,  256, 0, stream>>>(W_se2,   Wse2T,  128, 128, 128, 128, 1);
  pack_w_kernel<<<16,  256, 0, stream>>>(W_out,   WoutT,  256,   2, 256,  16, 1);

  const int blocks = (N + 15) / 16;
  diffusion_policy_kernel<<<blocks, TPB, SMEM_BYTES, stream>>>(
      states, W_se1, b_se1, b_se2, b_in, b_hid, b_scale, b_shift, b_out,
      WinT, WhidT, WscaleT, WshiftT, Wse2T, WoutT, nsp, out, N);
}
